// AureliusGATHetero_83846351552513
// MI455X (gfx1250) — matrix operand order repaired and checked by reading, not compiled, not run on hardware
//
#include <hip/hip_runtime.h>
#include <hip/hip_bf16.h>
#include <math.h>

typedef _Float16 half_t;
typedef __attribute__((ext_vector_type(16))) _Float16 v16h;
typedef __attribute__((ext_vector_type(8)))  _Float16 v8h;
typedef __attribute__((ext_vector_type(8)))  float    v8f;

#define HID 128
#define HEADS 4
#define PH 32
#define FIN 16
#define NEG_SLOPE 0.2f
#define BN_EPS 1e-5f

// ---------- ordered-float encoding for atomicMax on uint ----------
__device__ __forceinline__ unsigned f2ord(float f) {
    unsigned u = __float_as_uint(f);
    return (u & 0x80000000u) ? ~u : (u | 0x80000000u);
}
__device__ __forceinline__ float ord2f(unsigned u) {
    unsigned v = (u & 0x80000000u) ? (u & 0x7FFFFFFFu) : ~u;
    return __uint_as_float(v);
}

// ---------- weight convert + transpose: W[k][o] f32 -> Wt[o][k] f16 ----------
__global__ void k_convw(const float* __restrict__ W, half_t* __restrict__ Wt) {
    int t = blockIdx.x * blockDim.x + threadIdx.x;
    if (t >= HID * HID) return;
    int k = t >> 7, o = t & 127;
    Wt[o * HID + k] = (half_t)W[k * HID + o];
}

// ---------- input projection: h = elu(x @ Wp + bp) -> f16 ----------
__global__ void k_proj(const float* __restrict__ x, const float* __restrict__ W,
                       const float* __restrict__ b, half_t* __restrict__ h16, int n) {
    __shared__ float xs[FIN];
    int node = blockIdx.x;
    if (node >= n) return;
    int ch = threadIdx.x;                  // 128 threads
    if (ch < FIN) xs[ch] = x[node * FIN + ch];
    __syncthreads();
    float acc = b[ch];
#pragma unroll
    for (int i = 0; i < FIN; ++i) acc = fmaf(xs[i], W[i * HID + ch], acc);
    float e = acc > 0.f ? acc : (expf(acc) - 1.f);
    h16[node * HID + ch] = (half_t)e;
}

// ---------- WMMA A-fragment loader (ISA 16-bit A 16x32 layout) ----------
// lane holds row (lane&15); K chunks: [koff..koff+7] and [16+koff..16+koff+7], koff = lane<16?0:8
__device__ __forceinline__ v16h load_fragA(const half_t* __restrict__ p) {
    v8h lo = *(const v8h*)(p);        // K = koff + 0..7
    v8h hi = *(const v8h*)(p + 16);   // K = 16 + koff + 0..7
    v16h r;
#pragma unroll
    for (int i = 0; i < 8; ++i) { r[i] = lo[i]; r[i + 8] = hi[i]; }
    return r;
}

// ---------- WMMA B-fragment loader (ISA 16-bit B 32x16 layout) ----------
// lane holds column (lane&15); K chunks: [koff..koff+7] and [16+koff..16+koff+7], koff = lane<16?0:8
__device__ __forceinline__ v16h load_fragB(const half_t* __restrict__ p) {
    v8h lo = *(const v8h*)(p);        // K = koffB + 0..7
    v8h hi = *(const v8h*)(p + 16);   // K = 16 + koffB + 0..7
    v16h r;
#pragma unroll
    for (int i = 0; i < 8; ++i) { r[i] = lo[i]; r[i + 8] = hi[i]; }
    return r;
}

// ---------- dual GEMM via WMMA: XL = A@Wl, XR = A@Wr  (A: [N,128] f16, Bt: [2][128][128] f16, N-major) ----------
__global__ void k_gemm2(const half_t* __restrict__ A, const half_t* __restrict__ Bt,
                        float* __restrict__ XL, float* __restrict__ XR, int nrows) {
    int wave = (blockIdx.x * blockDim.x + threadIdx.x) >> 5;   // uniform per wave
    int lane = threadIdx.x & 31;
    int ntiles = nrows >> 4;
    if (wave >= ntiles) return;                                // whole-wave exit, EXEC stays full

    int lrow   = lane & 15;
    int khalfA = (lane < 16) ? 0 : 8;    // A: 8-interleaved K chunks
    int khalfB = (lane < 16) ? 0 : 8;    // B: 8-interleaved K chunks

    const half_t* arow = A + (size_t)(wave * 16 + lrow) * HID + khalfA;
    v16h afrag[4];
#pragma unroll
    for (int k = 0; k < 4; ++k) afrag[k] = load_fragA(arow + k * 32);

#pragma unroll
    for (int jt = 0; jt < 8; ++jt) {
        const half_t* b0 = Bt + (size_t)(jt * 16 + lrow) * HID + khalfB;
        const half_t* b1 = b0 + HID * HID;
        v8f acc0 = {}; v8f acc1 = {};
#pragma unroll
        for (int k = 0; k < 4; ++k) {
            v16h bf0 = load_fragB(b0 + k * 32);
            acc0 = __builtin_amdgcn_wmma_f32_16x16x32_f16(false, afrag[k], false, bf0,
                                                          (short)0, acc0, false, false);
            v16h bf1 = load_fragB(b1 + k * 32);
            acc1 = __builtin_amdgcn_wmma_f32_16x16x32_f16(false, afrag[k], false, bf1,
                                                          (short)0, acc1, false, false);
        }
        int col   = jt * 16 + lrow;
        int rbase = wave * 16 + ((lane < 16) ? 0 : 8);
#pragma unroll
        for (int r = 0; r < 8; ++r) {
            XL[(size_t)(rbase + r) * HID + col] = acc0[r];
            XR[(size_t)(rbase + r) * HID + col] = acc1[r];
        }
    }
}

// ---------- accumulator init with bias_fwd + bias_rev ----------
__global__ void k_init_acc(float* __restrict__ acc, const float* __restrict__ bf,
                           const float* __restrict__ br, int n) {
    int t = blockIdx.x * blockDim.x + threadIdx.x;
    if (t >= n * HID) return;
    int ch = t & 127;
    acc[t] = bf[ch] + br[ch];
}

// ---------- per-(node,head) max/sum init ----------
__global__ void k_init_ms(unsigned* __restrict__ m, float* __restrict__ s, int count) {
    int t = blockIdx.x * blockDim.x + threadIdx.x;
    if (t >= count) return;
    m[t] = 0x007FFFFFu;   // f2ord(-inf)
    s[t] = 0.f;
}

// ---------- attention logits + segment max ----------
__global__ void k_edge_logits(const int* __restrict__ src, const int* __restrict__ dst,
                              const float* __restrict__ xl, const float* __restrict__ xr,
                              const float* __restrict__ att,
                              float* __restrict__ ebuf, unsigned* __restrict__ mbuf, int ne) {
    int t = blockIdx.x * blockDim.x + threadIdx.x;
    if (t >= ne * HEADS) return;
    int h = t & 3, e = t >> 2;
    int s = src[e], d = dst[e];
    const float4* pl = (const float4*)(xl + (size_t)s * HID + h * PH);
    const float4* pr = (const float4*)(xr + (size_t)d * HID + h * PH);
    const float4* pa = (const float4*)(att + h * PH);
    float acc = 0.f;
#pragma unroll
    for (int c = 0; c < 8; ++c) {
        float4 a = pl[c], b = pr[c], w = pa[c];
        float v;
        v = a.x + b.x; v = v > 0.f ? v : NEG_SLOPE * v; acc = fmaf(w.x, v, acc);
        v = a.y + b.y; v = v > 0.f ? v : NEG_SLOPE * v; acc = fmaf(w.y, v, acc);
        v = a.z + b.z; v = v > 0.f ? v : NEG_SLOPE * v; acc = fmaf(w.z, v, acc);
        v = a.w + b.w; v = v > 0.f ? v : NEG_SLOPE * v; acc = fmaf(w.w, v, acc);
    }
    ebuf[t] = acc;
    atomicMax(mbuf + (d * HEADS + h), f2ord(acc));
}

// ---------- exp(logit - max) + segment sum ----------
__global__ void k_edge_exp(const int* __restrict__ dst, float* __restrict__ ebuf,
                           const unsigned* __restrict__ mbuf, float* __restrict__ sbuf, int ne) {
    int t = blockIdx.x * blockDim.x + threadIdx.x;
    if (t >= ne * HEADS) return;
    int h = t & 3, e = t >> 2;
    int d = dst[e];
    float m  = ord2f(mbuf[d * HEADS + h]);
    float ex = __expf(ebuf[t] - m);
    ebuf[t] = ex;
    atomicAdd(sbuf + d * HEADS + h, ex);
}

// ---------- alpha * xl[src] scatter-add into accumulator ----------
__global__ void k_edge_scatter(const int* __restrict__ src, const int* __restrict__ dst,
                               const float* __restrict__ xl, const float* __restrict__ ebuf,
                               const float* __restrict__ sbuf, float* __restrict__ acc, int ne) {
    int t = blockIdx.x * blockDim.x + threadIdx.x;
    if (t >= ne * HEADS) return;
    int h = t & 3, e = t >> 2;
    int s = src[e], d = dst[e];
    float alpha = ebuf[t] / (sbuf[d * HEADS + h] + 1e-16f);
    const float4* pl = (const float4*)(xl + (size_t)s * HID + h * PH);
    float* po = acc + (size_t)d * HID + h * PH;
#pragma unroll
    for (int c = 0; c < 8; ++c) {
        float4 v = pl[c];
        atomicAdd(po + 4 * c + 0, alpha * v.x);
        atomicAdd(po + 4 * c + 1, alpha * v.y);
        atomicAdd(po + 4 * c + 2, alpha * v.z);
        atomicAdd(po + 4 * c + 3, alpha * v.w);
    }
}

__global__ void k_zero_stats(float* __restrict__ stats) {
    int t = threadIdx.x;
    if (t < 256) stats[t] = 0.f;
}

// ---------- in-place ELU + per-channel sum / sumsq partials ----------
__global__ void k_elu_stats(float* __restrict__ acc, float* __restrict__ ssum,
                            float* __restrict__ ssq, int n) {
    int ch = threadIdx.x;     // 128 threads, one channel each
    float s = 0.f, s2 = 0.f;
    for (int r = blockIdx.x; r < n; r += gridDim.x) {
        float v = acc[(size_t)r * HID + ch];
        v = v > 0.f ? v : (expf(v) - 1.f);
        acc[(size_t)r * HID + ch] = v;
        s += v; s2 += v * v;
    }
    atomicAdd(ssum + ch, s);
    atomicAdd(ssq + ch, s2);
}

// ---------- batchnorm apply (biased var) + write h f16 for next layer ----------
__global__ void k_bn_apply(float* __restrict__ acc, const float* __restrict__ ssum,
                           const float* __restrict__ ssq, const float* __restrict__ gamma,
                           const float* __restrict__ beta, half_t* __restrict__ h16, int n) {
    int t = blockIdx.x * blockDim.x + threadIdx.x;
    if (t >= n * HID) return;
    int ch = t & 127;
    float inv_n = 1.f / (float)n;
    float mu  = ssum[ch] * inv_n;
    float var = ssq[ch] * inv_n - mu * mu;
    float v = (acc[t] - mu) * rsqrtf(var + BN_EPS) * gamma[ch] + beta[ch];
    acc[t] = v;
    h16[t] = (half_t)v;
}

// ---------- classifier: out[n,2] = h @ Wcls + bcls ----------
__global__ void k_cls(const float* __restrict__ h, const float* __restrict__ W,
                      const float* __restrict__ b, float* __restrict__ out, int n) {
    int node = blockIdx.x * blockDim.x + threadIdx.x;
    if (node >= n) return;
    float a0 = b[0], a1 = b[1];
    const float4* ph = (const float4*)(h + (size_t)node * HID);
#pragma unroll
    for (int c = 0; c < 32; ++c) {
        float4 v = ph[c];
        a0 = fmaf(v.x, W[(4 * c + 0) * 2 + 0], a0); a1 = fmaf(v.x, W[(4 * c + 0) * 2 + 1], a1);
        a0 = fmaf(v.y, W[(4 * c + 1) * 2 + 0], a0); a1 = fmaf(v.y, W[(4 * c + 1) * 2 + 1], a1);
        a0 = fmaf(v.z, W[(4 * c + 2) * 2 + 0], a0); a1 = fmaf(v.z, W[(4 * c + 2) * 2 + 1], a1);
        a0 = fmaf(v.w, W[(4 * c + 3) * 2 + 0], a0); a1 = fmaf(v.w, W[(4 * c + 3) * 2 + 1], a1);
    }
    out[node * 2 + 0] = a0;
    out[node * 2 + 1] = a1;
}

extern "C" void kernel_launch(void* const* d_in, const int* in_sizes, int n_in,
                              void* d_out, int out_size, void* d_ws, size_t ws_size,
                              hipStream_t stream) {
    (void)n_in; (void)out_size; (void)ws_size;

    // ---- map inputs (handles both insertion-order and jax alphabetical flatten) ----
    const float *x, *projW, *projb, *clsW, *clsb;
    const int *eif, *eir;
    const float *Wl[3][2], *Wr[3][2], *att[3][2], *bias[3][2], *gamma[3], *beta[3];
    int N, E;

    bool insertion = in_sizes[0] > in_sizes[1];   // x (N*16) first vs edge_index (2E) first
    if (insertion) {
        x = (const float*)d_in[0];
        eif = (const int*)d_in[1]; eir = (const int*)d_in[2];
        projW = (const float*)d_in[3]; projb = (const float*)d_in[4];
        for (int L = 0; L < 3; ++L) {
            int b0 = 5 + L * 10;
            Wl[L][0] = (const float*)d_in[b0 + 0]; Wr[L][0] = (const float*)d_in[b0 + 1];
            att[L][0] = (const float*)d_in[b0 + 2]; bias[L][0] = (const float*)d_in[b0 + 3];
            Wl[L][1] = (const float*)d_in[b0 + 4]; Wr[L][1] = (const float*)d_in[b0 + 5];
            att[L][1] = (const float*)d_in[b0 + 6]; bias[L][1] = (const float*)d_in[b0 + 7];
            gamma[L] = (const float*)d_in[b0 + 8]; beta[L] = (const float*)d_in[b0 + 9];
        }
        clsW = (const float*)d_in[35]; clsb = (const float*)d_in[36];
        N = in_sizes[0] / FIN; E = in_sizes[1] / 2;
    } else {
        eif = (const int*)d_in[0]; eir = (const int*)d_in[1];
        clsW = (const float*)d_in[2]; clsb = (const float*)d_in[3];
        for (int L = 0; L < 3; ++L) {
            int b0 = 4 + L * 10;
            beta[L] = (const float*)d_in[b0 + 0]; gamma[L] = (const float*)d_in[b0 + 1];
            Wl[L][0] = (const float*)d_in[b0 + 2]; Wr[L][0] = (const float*)d_in[b0 + 3];
            att[L][0] = (const float*)d_in[b0 + 4]; bias[L][0] = (const float*)d_in[b0 + 5];
            Wl[L][1] = (const float*)d_in[b0 + 6]; Wr[L][1] = (const float*)d_in[b0 + 7];
            att[L][1] = (const float*)d_in[b0 + 8]; bias[L][1] = (const float*)d_in[b0 + 9];
        }
        projW = (const float*)d_in[34]; projb = (const float*)d_in[35];
        x = (const float*)d_in[36];
        E = in_sizes[0] / 2; N = in_sizes[36] / FIN;
    }

    // ---- carve workspace (all 256B aligned) ----
    char* wsp = (char*)d_ws;
    size_t off = 0;
    auto carve = [&](size_t bytes) -> char* {
        char* p = wsp + off;
        off += (bytes + 255) & ~(size_t)255;
        return p;
    };
    half_t*   h16   = (half_t*)  carve((size_t)N * HID * sizeof(half_t));
    float*    accb  = (float*)   carve((size_t)N * HID * sizeof(float));
    float*    xlb   = (float*)   carve((size_t)N * HID * sizeof(float));
    float*    xrb   = (float*)   carve((size_t)N * HID * sizeof(float));
    half_t*   wbuf  = (half_t*)  carve((size_t)12 * HID * HID * sizeof(half_t));
    unsigned* mbuf  = (unsigned*)carve((size_t)N * HEADS * sizeof(unsigned));
    float*    sbuf  = (float*)   carve((size_t)N * HEADS * sizeof(float));
    float*    ebuf  = (float*)   carve((size_t)E * HEADS * sizeof(float));
    float*    stats = (float*)   carve(256 * sizeof(float));

    dim3 B256(256);

    // convert + transpose all 12 GAT weight matrices to f16 (N-major for WMMA B frags)
    for (int L = 0; L < 3; ++L)
        for (int T = 0; T < 2; ++T) {
            half_t* slot = wbuf + (size_t)((L * 2 + T) * 2) * HID * HID;
            k_convw<<<(HID * HID + 255) / 256, B256, 0, stream>>>(Wl[L][T], slot);
            k_convw<<<(HID * HID + 255) / 256, B256, 0, stream>>>(Wr[L][T], slot + HID * HID);
        }

    k_proj<<<N, 128, 0, stream>>>(x, projW, projb, h16, N);

    int gemmWaves  = N / 16;                               // N divisible by 16 (50000)
    int gemmBlocks = (gemmWaves * 32 + 255) / 256;
    int edgeBlocks = (E * HEADS + 255) / 256;
    int nmBlocks   = (N * HEADS + 255) / 256;
    int nhBlocks   = (N * HID + 255) / 256;

    for (int L = 0; L < 3; ++L) {
        k_init_acc<<<nhBlocks, B256, 0, stream>>>(accb, bias[L][0], bias[L][1], N);
        for (int T = 0; T < 2; ++T) {
            const int* ei = (T == 0) ? eif : eir;
            const half_t* wt = wbuf + (size_t)((L * 2 + T) * 2) * HID * HID;
            k_gemm2<<<gemmBlocks, B256, 0, stream>>>(h16, wt, xlb, xrb, N);
            k_init_ms<<<nmBlocks, B256, 0, stream>>>(mbuf, sbuf, N * HEADS);
            k_edge_logits<<<edgeBlocks, B256, 0, stream>>>(ei, ei + E, xlb, xrb, att[L][T],
                                                           ebuf, mbuf, E);
            k_edge_exp<<<edgeBlocks, B256, 0, stream>>>(ei + E, ebuf, mbuf, sbuf, E);
            k_edge_scatter<<<edgeBlocks, B256, 0, stream>>>(ei, ei + E, xlb, ebuf, sbuf, accb, E);
        }
        k_zero_stats<<<1, 256, 0, stream>>>(stats);
        k_elu_stats<<<512, 128, 0, stream>>>(accb, stats, stats + 128, N);
        k_bn_apply<<<nhBlocks, B256, 0, stream>>>(accb, stats, stats + 128,
                                                  gamma[L], beta[L], h16, N);
    }

    k_cls<<<(N + 255) / 256, B256, 0, stream>>>(accb, clsW, clsb, (float*)d_out, N);
}
